// SceneQueryGSHead_103079215856
// MI455X (gfx1250) — compile-verified, matrix operand order checked by reading, not run on hardware
//
// MI455X (gfx1250, CDNA5) implementation of SceneQueryGSHead forward.
// All GEMMs (transformer blocks, attention, MLP heads) run through a bf16
// V_WMMA_F32_16X16X32_BF16 tiled kernel: wave32, 8 waves/block, block tile
// 256x64, wave tile 32x64 (8 WMMAs per K-step), LDS double-buffered with
// CDNA5 async global->LDS b128 copies (s_wait_asynccnt). ~115 GFLOP of GEMM
// on an L2-resident (<100MB) working set -> bf16 matrix cores are the ceiling.
#include <hip/hip_runtime.h>
#include <stdint.h>

typedef __bf16 bf16_t;
typedef bf16_t v16bf __attribute__((ext_vector_type(16)));
typedef float  v8f   __attribute__((ext_vector_type(8)));
typedef unsigned int v4u __attribute__((ext_vector_type(4)));
typedef int v4i __attribute__((vector_size(16)));   // matches builtin param type

#define GF_GELU 1
#define GF_RES  2

static __device__ __forceinline__ bf16_t f2bf(float f) {
  union { float f; unsigned u; } a; a.f = f;
  unsigned r = a.u + 0x7FFFu + ((a.u >> 16) & 1u);
  union { unsigned short s; bf16_t b; } c; c.s = (unsigned short)(r >> 16);
  return c.b;
}
static __device__ __forceinline__ float gelu_f(float x) {
  float x3 = x * x * x;
  return 0.5f * x * (1.0f + tanhf(0.7978845608028654f * (x + 0.044715f * x3)));
}

// ---- CDNA5 async global->LDS copy (16B), guarded so compile never breaks ----
static __device__ __forceinline__ void cp_async16(bf16_t* lds, const bf16_t* gp) {
#if __has_builtin(__builtin_amdgcn_global_load_async_to_lds_b128)
  __builtin_amdgcn_global_load_async_to_lds_b128(
      (__attribute__((address_space(1))) v4i*)(bf16_t*)gp,
      (__attribute__((address_space(3))) v4i*)lds, 0, 0);
#elif __has_builtin(__builtin_amdgcn_load_to_lds)
  __builtin_amdgcn_load_to_lds((void*)gp,
      (__attribute__((address_space(3))) void*)lds, 16, 0, 0);
#else
  *(v4u*)lds = *(const v4u*)gp;
#endif
}
static __device__ __forceinline__ void wait_async() {
#if __has_builtin(__builtin_amdgcn_s_wait_asynccnt)
  __builtin_amdgcn_s_wait_asynccnt(0);
#elif defined(__gfx1250__)
  asm volatile("s_wait_asynccnt 0" ::: "memory");
#endif
}

// ===================== WMMA bf16 GEMM: C = act(A*B + bias) (+R) =============
struct GemmArgs {
  const bf16_t* A; const bf16_t* B; const float* bias; const float* R;
  float* Cf; bf16_t* Cb;
  int lda, ldb, ldr, ldc;
  int M, N, K;
  int transB, flags;
  long long sA, sB, sR, sC;
};

union Frag { v16bf v; v4u q[2]; };
union BU   { v4u q; bf16_t h[8]; };

__global__ __launch_bounds__(256) void wmma_gemm_kernel(GemmArgs g) {
  __shared__ __align__(16) bf16_t lA[2][256][40];   // A tile, 80B padded rows
  __shared__ __align__(16) bf16_t lBT[2][64][40];   // B tile, transposed [n][k]
  const int tid = threadIdx.x;
  const int lane = tid & 31, wv = tid >> 5;          // wave id = M sub-tile
  const int half = lane >> 4, l16 = lane & 15;
  const int m0 = blockIdx.y * 256, n0 = blockIdx.x * 64;
  const int z = blockIdx.z;
  const bf16_t* Ag = g.A + (size_t)z * g.sA;
  const bf16_t* Bg = g.B + (size_t)z * g.sB;
  const bf16_t bz = f2bf(0.f);

  v8f acc[2][4];
  for (int i = 0; i < 2; ++i) for (int j = 0; j < 4; ++j)
    for (int r = 0; r < 8; ++r) acc[i][j][r] = 0.f;

  const int nk = (g.K + 31) >> 5;

  auto loadA = [&](int buf, int k0) {
    bool fast = ((g.lda & 7) == 0) && (m0 + 256 <= g.M) && (k0 + 32 <= g.K) &&
                ((((size_t)Ag) & 15) == 0);
    if (fast) {
      for (int it = 0; it < 4; ++it) {               // 256 rows x 4 b128 chunks
        int chunk = tid + (it << 8);
        int row = chunk >> 2, kc = (chunk & 3) << 3;
        cp_async16(&lA[buf][row][kc], Ag + (size_t)(m0 + row) * g.lda + k0 + kc);
      }
    } else {
#pragma clang loop unroll(disable)
      for (int it = 0; it < 32; ++it) {
        int idx = tid + (it << 8);
        int row = idx >> 5, kk = idx & 31;
        int gm = m0 + row, gk = k0 + kk;
        lA[buf][row][kk] = (gm < g.M && gk < g.K) ? Ag[(size_t)gm * g.lda + gk] : bz;
      }
    }
  };
  auto loadB = [&](int buf, int k0) {
    bool inr = (n0 + 64 <= g.N) && (k0 + 32 <= g.K) && ((g.ldb & 7) == 0) &&
               ((((size_t)Bg) & 15) == 0);
    if (inr && g.transB) {
      // lBT[n][k] mirrors row-major B[n][k]: pure async b128 copies
      int row = tid >> 2, kc = (tid & 3) << 3;       // 64 rows x 4 chunks
      cp_async16(&lBT[buf][row][kc], Bg + (size_t)(n0 + row) * g.ldb + k0 + kc);
    } else if (inr) {
      // one b128 row chunk per thread, transpose via 8 ds_store_b16
      int k = tid >> 3, nc = (tid & 7) << 3;         // 32 k x 8 n-chunks
      BU u; u.q = *(const v4u*)(Bg + (size_t)(k0 + k) * g.ldb + n0 + nc);
      for (int j = 0; j < 8; ++j) lBT[buf][nc + j][k] = u.h[j];
    } else {
#pragma clang loop unroll(disable)
      for (int it = 0; it < 8; ++it) {
        int idx = tid + (it << 8);
        int n = idx & 63, kk = idx >> 6;
        int gn = n0 + n, gk = k0 + kk;
        bf16_t val = bz;
        if (gn < g.N && gk < g.K)
          val = g.transB ? Bg[(size_t)gn * g.ldb + gk] : Bg[(size_t)gk * g.ldb + gn];
        lBT[buf][n][kk] = val;
      }
    }
  };

  loadA(0, 0); loadB(0, 0);
  wait_async();
  __syncthreads();
  for (int kt = 0; kt < nk; ++kt) {
    const int cbuf = kt & 1, nbuf = cbuf ^ 1;
    if (kt + 1 < nk) { loadA(nbuf, (kt + 1) << 5); loadB(nbuf, (kt + 1) << 5); }
    Frag af[2], bf_[4];
    // 16-bit A/B fragment layout (ISA 7.12.2): lane half picks K +8 / +24
    for (int i = 0; i < 2; ++i) {
      const bf16_t* ap = &lA[cbuf][wv * 32 + i * 16 + l16][half * 8];
      af[i].q[0] = *(const v4u*)(ap);
      af[i].q[1] = *(const v4u*)(ap + 16);
    }
    for (int j = 0; j < 4; ++j) {
      const bf16_t* bp = &lBT[cbuf][j * 16 + l16][half * 8];
      bf_[j].q[0] = *(const v4u*)(bp);
      bf_[j].q[1] = *(const v4u*)(bp + 16);
    }
    for (int i = 0; i < 2; ++i)
      for (int j = 0; j < 4; ++j)
        acc[i][j] = __builtin_amdgcn_wmma_f32_16x16x32_bf16(
            false, af[i].v, false, bf_[j].v, (short)0, acc[i][j], false, false);
    wait_async();
    __syncthreads();
  }

  const float* Rg = g.R ? g.R + (size_t)z * g.sR : nullptr;
  float* Cfz = g.Cf ? g.Cf + (size_t)z * g.sC : nullptr;
  bf16_t* Cbz = g.Cb ? g.Cb + (size_t)z * g.sC : nullptr;
  for (int i = 0; i < 2; ++i)
    for (int j = 0; j < 4; ++j)
      for (int r = 0; r < 8; ++r) {
        int m = m0 + wv * 32 + i * 16 + half * 8 + r;
        int n = n0 + j * 16 + l16;
        if (m < g.M && n < g.N) {
          float v = acc[i][j][r];
          if (g.bias) v += g.bias[n];
          if (g.flags & GF_GELU) v = gelu_f(v);
          if (Rg) v += Rg[(size_t)m * g.ldr + n];
          if (Cfz) Cfz[(size_t)m * g.ldc + n] = v;
          if (Cbz) Cbz[(size_t)m * g.ldc + n] = f2bf(v);
        }
      }
}

// ============================ glue kernels ==================================
__global__ void pack_bf16_kernel(const float* __restrict__ s, bf16_t* __restrict__ d, int n) {
  int i = blockIdx.x * blockDim.x + threadIdx.x;
  if (i < n) d[i] = f2bf(s[i]);
}

__global__ void layernorm_kernel(const float* __restrict__ x, const float* __restrict__ g,
                                 const float* __restrict__ b, float* yf, bf16_t* yb,
                                 int Dm, int do_gelu) {
  const int row = blockIdx.x;
  const float* xr = x + (size_t)row * Dm;
  __shared__ float red[8], red2[8], bc[2];
  const int tid = threadIdx.x, lane = tid & 31, w = tid >> 5;
  float s = 0.f, s2 = 0.f;
  for (int i = tid; i < Dm; i += blockDim.x) { float v = xr[i]; s += v; s2 += v * v; }
  for (int o = 16; o; o >>= 1) { s += __shfl_xor(s, o, 32); s2 += __shfl_xor(s2, o, 32); }
  if (lane == 0) { red[w] = s; red2[w] = s2; }
  __syncthreads();
  if (tid == 0) {
    float ts = 0.f, ts2 = 0.f;
    for (int i = 0; i < (int)(blockDim.x >> 5); ++i) { ts += red[i]; ts2 += red2[i]; }
    float mean = ts / Dm;
    bc[0] = mean; bc[1] = rsqrtf(ts2 / Dm - mean * mean + 1e-5f);
  }
  __syncthreads();
  float mean = bc[0], inv = bc[1];
  for (int i = tid; i < Dm; i += blockDim.x) {
    float v = (xr[i] - mean) * inv * g[i] + b[i];
    if (do_gelu) v = gelu_f(v);
    if (yf) yf[(size_t)row * Dm + i] = v;
    if (yb) yb[(size_t)row * Dm + i] = f2bf(v);
  }
}

__global__ void softmax_kernel(const float* __restrict__ X, bf16_t* __restrict__ P,
                               int Sk, float scale) {
  const int row = blockIdx.x;
  const float* xr = X + (size_t)row * Sk;
  bf16_t* pr = P + (size_t)row * Sk;
  __shared__ float red[8], bc[2];
  const int tid = threadIdx.x, lane = tid & 31, w = tid >> 5;
  float mx = -3.4e38f;
  for (int i = tid; i < Sk; i += blockDim.x) mx = fmaxf(mx, xr[i] * scale);
  for (int o = 16; o; o >>= 1) mx = fmaxf(mx, __shfl_xor(mx, o, 32));
  if (lane == 0) red[w] = mx;
  __syncthreads();
  if (tid == 0) {
    float m = -3.4e38f;
    for (int i = 0; i < (int)(blockDim.x >> 5); ++i) m = fmaxf(m, red[i]);
    bc[0] = m;
  }
  __syncthreads();
  float rm = bc[0], s = 0.f;
  for (int i = tid; i < Sk; i += blockDim.x) s += expf(xr[i] * scale - rm);
  for (int o = 16; o; o >>= 1) s += __shfl_xor(s, o, 32);
  if (lane == 0) red[w] = s;
  __syncthreads();
  if (tid == 0) {
    float t = 0.f;
    for (int i = 0; i < (int)(blockDim.x >> 5); ++i) t += red[i];
    bc[1] = 1.f / t;
  }
  __syncthreads();
  float inv = bc[1];
  for (int i = tid; i < Sk; i += blockDim.x)
    pr[i] = f2bf(expf(xr[i] * scale - rm) * inv);
}

__global__ void gather_kernel(const float* __restrict__ pts, float* __restrict__ o,
                              int n, int stride, int off, int P) {
  int i = blockIdx.x * blockDim.x + threadIdx.x;
  if (i >= n) return;
  int idx = (i * stride + off) % P;
  o[i * 3 + 0] = pts[idx * 3 + 0];
  o[i * 3 + 1] = pts[idx * 3 + 1];
  o[i * 3 + 2] = pts[idx * 3 + 2];
}

__global__ void minmax_kernel(const float* __restrict__ pts, int P, float* __restrict__ part) {
  __shared__ float sh[256];
  const int tid = threadIdx.x;
  float mn[3] = {3.4e38f, 3.4e38f, 3.4e38f}, mx[3] = {-3.4e38f, -3.4e38f, -3.4e38f};
  for (int i = blockIdx.x * 256 + tid; i < P; i += gridDim.x * 256)
    for (int c = 0; c < 3; ++c) {
      float v = pts[i * 3 + c];
      mn[c] = fminf(mn[c], v); mx[c] = fmaxf(mx[c], v);
    }
  for (int c = 0; c < 3; ++c) {
    sh[tid] = mn[c]; __syncthreads();
    for (int s = 128; s; s >>= 1) { if (tid < s) sh[tid] = fminf(sh[tid], sh[tid + s]); __syncthreads(); }
    if (tid == 0) part[blockIdx.x * 6 + c] = sh[0];
    __syncthreads();
    sh[tid] = mx[c]; __syncthreads();
    for (int s = 128; s; s >>= 1) { if (tid < s) sh[tid] = fmaxf(sh[tid], sh[tid + s]); __syncthreads(); }
    if (tid == 0) part[blockIdx.x * 6 + 3 + c] = sh[0];
    __syncthreads();
  }
}

// scalars: [0..2]=center [3]=scale [4]=log(scale) [5]=coarse_vs [6]=fine_vs
__global__ void scalars_kernel(const float* __restrict__ part, int nparts,
                               const float* __restrict__ cpts, float* __restrict__ sc) {
  const int tid = threadIdx.x;  // 512 threads
  __shared__ float sh[512], smin[3], smax[3], scen[3], slohi[2];
  __shared__ int scnt;
  for (int c = 0; c < 3; ++c) {
    float mn = 3.4e38f, mx = -3.4e38f;
    for (int i = tid; i < nparts; i += 512) {
      mn = fminf(mn, part[i * 6 + c]); mx = fmaxf(mx, part[i * 6 + 3 + c]);
    }
    sh[tid] = mn; __syncthreads();
    for (int s = 256; s; s >>= 1) { if (tid < s) sh[tid] = fminf(sh[tid], sh[tid + s]); __syncthreads(); }
    if (tid == 0) smin[c] = sh[0];
    __syncthreads();
    sh[tid] = mx; __syncthreads();
    for (int s = 256; s; s >>= 1) { if (tid < s) sh[tid] = fmaxf(sh[tid], sh[tid + s]); __syncthreads(); }
    if (tid == 0) smax[c] = sh[0];
    __syncthreads();
  }
  for (int c = 0; c < 3; ++c) {
    sh[tid] = cpts[tid * 3 + c]; __syncthreads();
    for (int s = 256; s; s >>= 1) { if (tid < s) sh[tid] += sh[tid + s]; __syncthreads(); }
    if (tid == 0) scen[c] = sh[0] / 512.f;
    __syncthreads();
  }
  float dx = cpts[tid * 3 + 0] - scen[0], dy = cpts[tid * 3 + 1] - scen[1],
        dz = cpts[tid * 3 + 2] - scen[2];
  float dist = sqrtf(dx * dx + dy * dy + dz * dz);
  sh[tid] = dist; __syncthreads();
  for (int s = 256; s; s >>= 1) { if (tid < s) sh[tid] = fmaxf(sh[tid], sh[tid + s]); __syncthreads(); }
  if (tid == 0) { slohi[0] = 0.f; slohi[1] = sh[0]; }
  __syncthreads();
#pragma clang loop unroll(disable)
  for (int it = 0; it < 30; ++it) {      // binary-search 0.99 quantile
    if (tid == 0) scnt = 0;
    __syncthreads();
    float mid = 0.5f * (slohi[0] + slohi[1]);
    if (dist <= mid) atomicAdd(&scnt, 1);
    __syncthreads();
    if (tid == 0) { if (scnt >= 507) slohi[1] = mid; else slohi[0] = mid; }
    __syncthreads();
  }
  if (tid == 0) {
    float scale = fmaxf(0.5f * (slohi[0] + slohi[1]), 1e-8f);
    float d2 = 0.f;
    for (int c = 0; c < 3; ++c) { float d = smax[c] - smin[c]; d2 += d * d; }
    float diag = sqrtf(d2);
    sc[0] = scen[0]; sc[1] = scen[1]; sc[2] = scen[2];
    sc[3] = scale; sc[4] = logf(scale);
    sc[5] = diag / cbrtf(512.f); sc[6] = diag / cbrtf(1536.f);
  }
}

__global__ void fourier_kernel(const float* __restrict__ pts, const float* __restrict__ sc,
                               bf16_t* __restrict__ o, int N) {
  int n = blockIdx.x * blockDim.x + threadIdx.x;
  if (n >= N) return;
  const float inv = 1.f / sc[3];
  float x[3];
  for (int c = 0; c < 3; ++c) x[c] = (pts[n * 3 + c] - sc[c]) * inv;
  bf16_t* op = o + (size_t)n * 39;
  for (int c = 0; c < 3; ++c) op[c] = f2bf(x[c]);
  float fb = 3.14159265358979323846f;
  for (int f = 0; f < 6; ++f) {
    for (int c = 0; c < 3; ++c) {
      float xi = x[c] * fb;
      op[3 + c * 6 + f] = f2bf(sinf(xi));
      op[21 + c * 6 + f] = f2bf(cosf(xi));
    }
    fb *= 2.f;
  }
}

__global__ void project_kernel(const float* __restrict__ pts, const float* __restrict__ E,
                               const float* __restrict__ Km, float* __restrict__ U,
                               float* __restrict__ Vv, int* __restrict__ Vis, int N) {
  int i = blockIdx.x * blockDim.x + threadIdx.x;
  if (i >= 4 * N) return;
  int view = i / N, n = i % N;
  const float* e = E + view * 16;
  const float* km = Km + view * 9;
  float p0 = pts[n * 3 + 0], p1 = pts[n * 3 + 1], p2 = pts[n * 3 + 2];
  float cx = e[0] * p0 + e[1] * p1 + e[2] * p2 + e[3];
  float cy = e[4] * p0 + e[5] * p1 + e[6] * p2 + e[7];
  float cz = e[8] * p0 + e[9] * p1 + e[10] * p2 + e[11];
  float ux = km[0] * cx + km[1] * cy + km[2] * cz;
  float uy = km[3] * cx + km[4] * cy + km[5] * cz;
  float uz = km[6] * cx + km[7] * cy + km[8] * cz;
  float zs = (fabsf(uz) > 1e-6f) ? uz : 1e-6f;
  float u = ux / zs, v = uy / zs;
  U[i] = u; Vv[i] = v;
  Vis[i] = (uz > 0.001f) && (u >= 0.f) && (u <= 223.f) && (v >= 0.f) && (v <= 223.f);
}

static __device__ __forceinline__ float bilin(const float* fc, float u, float v) {
  u = fminf(fmaxf(u, 0.f), 223.f);
  v = fminf(fmaxf(v, 0.f), 223.f);
  int u0 = (int)floorf(u), v0 = (int)floorf(v);
  int u1 = min(u0 + 1, 223), v1 = min(v0 + 1, 223);
  float du = u - u0, dv = v - v0;
  return fc[v0 * 224 + u0] * (1 - du) * (1 - dv) + fc[v0 * 224 + u1] * du * (1 - dv) +
         fc[v1 * 224 + u0] * (1 - du) * dv + fc[v1 * 224 + u1] * du * dv;
}

__global__ void bilinear_kernel(const float* __restrict__ feat, const float* __restrict__ U,
                                const float* __restrict__ Vv, float* __restrict__ sf,
                                bf16_t* __restrict__ sb, int N) {
  int n = blockIdx.x, view = blockIdx.y, c = threadIdx.x;   // 128 threads
  const float* fc = feat + ((size_t)view * 128 + c) * 50176;
  float val = bilin(fc, U[view * N + n], Vv[view * N + n]);
  size_t o = ((size_t)n * 4 + view) * 128 + c;
  sf[o] = val; sb[o] = f2bf(val);
}

__global__ void slot_sample_kernel(const float* __restrict__ feat, const float* __restrict__ U,
                                   const float* __restrict__ Vv, const int* __restrict__ Vis,
                                   const float* __restrict__ slot_uv, bf16_t* __restrict__ sb,
                                   int N, int row_base) {
  int n = blockIdx.x, k = blockIdx.y, c = threadIdx.x;   // 128 threads
  float du0 = slot_uv[k * 2 + 0] * 2.0f, dv0 = slot_uv[k * 2 + 1] * 2.0f;
  float acc = 0.f, cnt = 0.f;
  for (int view = 0; view < 4; ++view) {
    float m = Vis[view * N + n] ? 1.f : 0.f;
    const float* fc = feat + ((size_t)view * 128 + c) * 50176;
    acc += m * bilin(fc, U[view * N + n] + du0, Vv[view * N + n] + dv0);
    cnt += m;
  }
  size_t r = (size_t)(row_base + n * 4 + k);
  sb[r * 128 + c] = f2bf(acc / fmaxf(cnt, 1.f));
}

__global__ void add_q_pack_kernel(const float* __restrict__ proj, const float* __restrict__ q,
                                  int ldq, bf16_t* __restrict__ o, long total) {
  long i = (long)blockIdx.x * blockDim.x + threadIdx.x;
  if (i >= total) return;
  int c = (int)(i % 256);
  long n = i / 1024;
  o[i] = f2bf(proj[i] + q[(size_t)n * ldq + c]);
}

__global__ void dot_kernel(const float* __restrict__ X, const float* __restrict__ w,
                           const float* __restrict__ b, float* __restrict__ o, int rows, int K) {
  int r = blockIdx.x * blockDim.x + threadIdx.x;
  if (r >= rows) return;
  float s = b[0];
  for (int k = 0; k < K; ++k) s += X[(size_t)r * K + k] * w[k];
  o[r] = s;
}

__global__ void aggregate_kernel(const float* __restrict__ logits, const int* __restrict__ Vis,
                                 const float* __restrict__ proj, float* __restrict__ op,
                                 int N, int ldo) {
  int n = blockIdx.x, c = threadIdx.x;   // 256 threads
  float lg[4], mx = -3.4e38f;
  for (int v = 0; v < 4; ++v) {
    lg[v] = logits[n * 4 + v] + (Vis[v * N + n] ? 0.f : -1e9f);
    mx = fmaxf(mx, lg[v]);
  }
  float se = 0.f;
  for (int v = 0; v < 4; ++v) { lg[v] = expf(lg[v] - mx); se += lg[v]; }
  float acc = 0.f;
  for (int v = 0; v < 4; ++v)
    acc += (lg[v] / se) * proj[((size_t)n * 4 + v) * 256 + c];
  op[(size_t)n * ldo + c] = acc;
}

__global__ void slotfeat_kernel(const float* __restrict__ qlat, const float* __restrict__ tok,
                                const float* __restrict__ s2d, bf16_t* __restrict__ o, long total) {
  long i = (long)blockIdx.x * blockDim.x + threadIdx.x;
  if (i >= total) return;
  int c = (int)(i % 128);
  long r = i / 128;
  long n = r / 4; int k = (int)(r % 4);
  float v = (c < 64) ? qlat[n * 64 + c] : tok[k * 64 + (c - 64)] + s2d[r * 64 + (c - 64)];
  o[i] = f2bf(v);
}

__global__ void gsp_adjust_kernel(float* __restrict__ gsp, const float* __restrict__ sc, int rows) {
  int r = blockIdx.x * blockDim.x + threadIdx.x;
  if (r >= rows) return;
  float ls = sc[4];
  gsp[(size_t)r * 85 + 1] += ls; gsp[(size_t)r * 85 + 2] += ls; gsp[(size_t)r * 85 + 3] += ls;
}

__global__ void offsets_kernel(const float* __restrict__ offv, const float* __restrict__ ap,
                               const float* __restrict__ sc, float* __restrict__ gpos, int rows) {
  int i = blockIdx.x * blockDim.x + threadIdx.x;
  if (i >= rows * 3) return;
  int r = i / 3, c = i % 3;
  int n = r / 4;
  float vs = (n < 512) ? sc[5] : sc[6];
  gpos[i] = ap[n * 3 + c] + tanhf(offv[i]) * vs;
}

// ============================ host orchestration ============================
// Param leaf indices assuming insertion-order flattening of setup_inputs():
enum {
  IN_PTS = 0, IN_FEAT, IN_EXT, IN_KM,
  P_POS_W1 = 4, P_POS_B1, P_POS_W2, P_POS_B2,
  P_FEAT_W, P_FEAT_B, P_AGG_W1, P_AGG_B1, P_AGG_W2, P_AGG_B2,
  P_BLK0 = 14,                 // 8 blocks x 16 leaves (wq,bq,wk,bk,wv,bv,wo,bo,mw1,mb1,mw2,mb2,ln1g,ln1b,ln2g,ln2b)
  P_LAT_W1 = 142, P_LAT_B1, P_LAT_LN_G, P_LAT_LN_B, P_LAT_W2, P_LAT_B2,
  P_LN_G = 148, P_LN_B,
  P_OUT_W1 = 150, P_OUT_B1, P_OUT_W2, P_OUT_B2,
  P_OFF_W1 = 154, P_OFF_B1, P_OFF_W2, P_OFF_B2,
  P_SLOT_TOKENS = 158, P_SLOT_UV, P_SLOT_W, P_SLOT_B
};

extern "C" void kernel_launch(void* const* d_in, const int* in_sizes, int n_in,
                              void* d_out, int out_size, void* d_ws, size_t ws_size,
                              hipStream_t stream) {
  (void)in_sizes; (void)n_in; (void)out_size; (void)ws_size;
  auto F = [&](int i) { return (const float*)d_in[i]; };
  const float* pts  = F(IN_PTS);
  const float* feat = F(IN_FEAT);
  const float* E    = F(IN_EXT);
  const float* Km   = F(IN_KM);
  const int P = 4 * 224 * 224;
  float* out = (float*)d_out;
  float* gsp_out  = out;
  float* gpos_out = out + 8192 * 85;
  float* qlat_out = out + 8192 * 85 + 8192 * 3;

  char* wp = (char*)d_ws;
  auto alloc = [&](size_t bytes) {
    void* p = (void*)wp; wp += (bytes + 255) & ~(size_t)255; return p;
  };
  float*  partial = (float*)alloc(1024 * 6 * 4);
  float*  scalars = (float*)alloc(64 * 4);
  float*  all_pts = (float*)alloc(2048 * 3 * 4);
  float*  cu = (float*)alloc(4 * 2048 * 4);
  float*  cvv = (float*)alloc(4 * 2048 * 4);
  int*    cvis = (int*)alloc(4 * 2048 * 4);
  float*  fu = (float*)alloc(4 * 2048 * 4);
  float*  fvv = (float*)alloc(4 * 2048 * 4);
  int*    fvis = (int*)alloc(4 * 2048 * 4);
  bf16_t* fourb = (bf16_t*)alloc((size_t)2048 * 39 * 2);
  bf16_t* wslot = (bf16_t*)alloc((size_t)2048 * 512 * 2);
  bf16_t* t1b   = (bf16_t*)alloc((size_t)2048 * 256 * 2);
  float*  xfull = (float*)alloc((size_t)2048 * 512 * 4);
  float*  samp_f = (float*)alloc((size_t)8192 * 128 * 4);
  bf16_t* samp_b = (bf16_t*)alloc((size_t)8192 * 128 * 2);
  float*  projf = (float*)alloc((size_t)8192 * 256 * 4);
  bf16_t* abb   = (bf16_t*)alloc((size_t)8192 * 256 * 2);
  float*  t2f   = (float*)alloc((size_t)8192 * 128 * 4);
  float*  logits = (float*)alloc(8192 * 4);
  bf16_t* hb  = (bf16_t*)alloc((size_t)2048 * 512 * 2);
  bf16_t* cb2 = (bf16_t*)alloc((size_t)512 * 512 * 2);
  bf16_t* qb  = (bf16_t*)alloc((size_t)2048 * 512 * 2);
  bf16_t* kb  = (bf16_t*)alloc((size_t)2048 * 512 * 2);
  bf16_t* vb  = (bf16_t*)alloc((size_t)2048 * 512 * 2);
  bf16_t* ob  = (bf16_t*)alloc((size_t)2048 * 512 * 2);
  float*  x2  = (float*)alloc((size_t)2048 * 512 * 4);
  bf16_t* tb  = (bf16_t*)alloc((size_t)2048 * 2048 * 2);
  float*  attf = (float*)alloc((size_t)2048 * 2048 * 4);
  bf16_t* pb   = (bf16_t*)alloc((size_t)2048 * 2048 * 2);
  float*  l1   = (float*)alloc((size_t)2048 * 256 * 4);
  bf16_t* l1b  = (bf16_t*)alloc((size_t)2048 * 256 * 2);
  float*  latf = (float*)alloc((size_t)2048 * 64 * 4);
  bf16_t* sfb  = (bf16_t*)alloc((size_t)8192 * 128 * 2);
  bf16_t* o1b  = (bf16_t*)alloc((size_t)8192 * 128 * 2);
  bf16_t* off1b = (bf16_t*)alloc((size_t)8192 * 64 * 2);
  float*  offv  = (float*)alloc((size_t)8192 * 3 * 4);
  float*  s2d   = (float*)alloc((size_t)8192 * 64 * 4);

  auto gemm = [&](const bf16_t* A, int lda, const bf16_t* B, int ldb, int transB,
                  const float* bias, const float* R, int ldr,
                  float* Cf, bf16_t* Cb, int ldc, int M, int N, int K, int flags) {
    GemmArgs g;
    g.A = A; g.B = B; g.bias = bias; g.R = R; g.Cf = Cf; g.Cb = Cb;
    g.lda = lda; g.ldb = ldb; g.ldr = ldr; g.ldc = ldc;
    g.M = M; g.N = N; g.K = K; g.transB = transB; g.flags = flags;
    g.sA = 0; g.sB = 0; g.sR = 0; g.sC = 0;
    dim3 grid((N + 63) / 64, (M + 255) / 256, 1);
    wmma_gemm_kernel<<<grid, 256, 0, stream>>>(g);
  };
  auto PW = [&](int idx, long n) {  // pack an f32 weight into the bf16 slot
    pack_bf16_kernel<<<(int)((n + 255) / 256), 256, 0, stream>>>(F(idx), wslot, (int)n);
  };

  // ---- voxelize + scene scalars ----
  gather_kernel<<<2, 256, 0, stream>>>(pts, all_pts, 512, P / 512, 0, P);
  gather_kernel<<<6, 256, 0, stream>>>(pts, all_pts + 512 * 3, 1536, P / 1536, 7, P);
  minmax_kernel<<<512, 256, 0, stream>>>(pts, P, partial);
  scalars_kernel<<<1, 512, 0, stream>>>(partial, 512, all_pts, scalars);

  // ---- per-branch feature extraction (pos-MLP + multiview aggregation) ----
  auto run_branch = [&](const float* bpts, int N, float* bout,
                        float* U, float* Vv, int* Vis) {
    fourier_kernel<<<(N + 255) / 256, 256, 0, stream>>>(bpts, scalars, fourb, N);
    PW(P_POS_W1, 39 * 256);
    gemm(fourb, 39, wslot, 256, 0, F(P_POS_B1), nullptr, 0, nullptr, t1b, 256, N, 256, 39, GF_GELU);
    PW(P_POS_W2, 256 * 256);
    gemm(t1b, 256, wslot, 256, 0, F(P_POS_B2), nullptr, 0, bout, nullptr, 512, N, 256, 256, 0);
    project_kernel<<<(4 * N + 255) / 256, 256, 0, stream>>>(bpts, E, Km, U, Vv, Vis, N);
    dim3 bg(N, 4);
    bilinear_kernel<<<bg, 128, 0, stream>>>(feat, U, Vv, samp_f, samp_b, N);
    PW(P_FEAT_W, 128 * 256);
    gemm(samp_b, 128, wslot, 256, 0, F(P_FEAT_B), nullptr, 0, projf, nullptr, 256, N * 4, 256, 128, 0);
    long tot = (long)N * 1024;
    add_q_pack_kernel<<<(int)((tot + 255) / 256), 256, 0, stream>>>(projf, bout, 512, abb, tot);
    PW(P_AGG_W1, 256 * 128);
    gemm(abb, 256, wslot, 128, 0, F(P_AGG_B1), nullptr, 0, t2f, nullptr, 128, N * 4, 128, 256, GF_GELU);
    dot_kernel<<<(N * 4 + 255) / 256, 256, 0, stream>>>(t2f, F(P_AGG_W2), F(P_AGG_B2), logits, N * 4, 128);
    aggregate_kernel<<<N, 256, 0, stream>>>(logits, Vis, projf, bout + 256, N, 512);
  };
  run_branch(all_pts, 512, xfull, cu, cvv, cvis);
  run_branch(all_pts + 512 * 3, 1536, xfull + (size_t)512 * 512, fu, fvv, fvis);

  // ---- transformer blocks (all matmuls via bf16 WMMA) ----
  auto run_block = [&](int base, float* x, int Sq, const float* ctx, int Sk) {
    layernorm_kernel<<<Sq, 256, 0, stream>>>(x, F(base + 12), F(base + 13), nullptr, hb, 512, 0);
    bf16_t* ctxb = hb;
    if (ctx != (const float*)x) {
      layernorm_kernel<<<Sk, 256, 0, stream>>>(ctx, F(base + 12), F(base + 13), nullptr, cb2, 512, 0);
      ctxb = cb2;
    }
    PW(base + 0, 512 * 512);
    gemm(hb, 512, wslot, 512, 0, F(base + 1), nullptr, 0, nullptr, qb, 512, Sq, 512, 512, 0);
    PW(base + 2, 512 * 512);
    gemm(ctxb, 512, wslot, 512, 0, F(base + 3), nullptr, 0, nullptr, kb, 512, Sk, 512, 512, 0);
    PW(base + 4, 512 * 512);
    gemm(ctxb, 512, wslot, 512, 0, F(base + 5), nullptr, 0, nullptr, vb, 512, Sk, 512, 512, 0);
    for (int h = 0; h < 8; ++h) {   // per-head: QK^T -> softmax -> PV
      gemm(qb + h * 64, 512, kb + h * 64, 512, 1, nullptr, nullptr, 0,
           attf, nullptr, Sk, Sq, Sk, 64, 0);
      softmax_kernel<<<Sq, 256, 0, stream>>>(attf, pb, Sk, 0.125f);
      gemm(pb, Sk, vb + h * 64, 512, 0, nullptr, nullptr, 0,
           nullptr, ob + h * 64, 512, Sq, 64, Sk, 0);
    }
    PW(base + 6, 512 * 512);
    gemm(ob, 512, wslot, 512, 0, F(base + 7), x, 512, x2, nullptr, 512, Sq, 512, 512, GF_RES);
    layernorm_kernel<<<Sq, 256, 0, stream>>>(x2, F(base + 14), F(base + 15), nullptr, hb, 512, 0);
    PW(base + 8, 512 * 2048);
    gemm(hb, 512, wslot, 2048, 0, F(base + 9), nullptr, 0, nullptr, tb, 2048, Sq, 2048, 512, GF_GELU);
    PW(base + 10, 2048 * 512);
    gemm(tb, 2048, wslot, 512, 0, F(base + 11), x2, 512, x, nullptr, 512, Sq, 512, 2048, GF_RES);
  };
  run_block(P_BLK0 + 0,  xfull, 512, xfull, 512);                                  // coarse
  run_block(P_BLK0 + 16, xfull, 512, xfull, 512);
  run_block(P_BLK0 + 32, xfull + (size_t)512 * 512, 1536, xfull, 512);             // cross
  run_block(P_BLK0 + 48, xfull + (size_t)512 * 512, 1536, xfull, 512);
  for (int b = 0; b < 4; ++b)                                                       // self
    run_block(P_BLK0 + 64 + 16 * b, xfull, 2048, xfull, 2048);

  // ---- latent head: qlat = LN(gelu(LN(full@W1+b1))@W2+b2) ----
  pack_bf16_kernel<<<(2048 * 512 + 255) / 256, 256, 0, stream>>>(xfull, hb, 2048 * 512);
  PW(P_LAT_W1, 512 * 256);
  gemm(hb, 512, wslot, 256, 0, F(P_LAT_B1), nullptr, 0, l1, nullptr, 256, 2048, 256, 512, 0);
  layernorm_kernel<<<2048, 256, 0, stream>>>(l1, F(P_LAT_LN_G), F(P_LAT_LN_B), nullptr, l1b, 256, 1);
  PW(P_LAT_W2, 256 * 64);
  gemm(l1b, 256, wslot, 64, 0, F(P_LAT_B2), nullptr, 0, latf, nullptr, 64, 2048, 64, 256, 0);
  layernorm_kernel<<<2048, 256, 0, stream>>>(latf, F(P_LN_G), F(P_LN_B), qlat_out, nullptr, 64, 0);

  // ---- slot features + output heads ----
  dim3 sg1(512, 4), sg2(1536, 4);
  slot_sample_kernel<<<sg1, 128, 0, stream>>>(feat, cu, cvv, cvis, F(P_SLOT_UV), samp_b, 512, 0);
  slot_sample_kernel<<<sg2, 128, 0, stream>>>(feat, fu, fvv, fvis, F(P_SLOT_UV), samp_b, 1536, 2048);
  PW(P_SLOT_W, 128 * 64);
  gemm(samp_b, 128, wslot, 64, 0, F(P_SLOT_B), nullptr, 0, s2d, nullptr, 64, 8192, 64, 128, 0);
  slotfeat_kernel<<<(8192 * 128 + 255) / 256, 256, 0, stream>>>(
      qlat_out, F(P_SLOT_TOKENS), s2d, sfb, (long)8192 * 128);
  PW(P_OUT_W1, 128 * 128);
  gemm(sfb, 128, wslot, 128, 0, F(P_OUT_B1), nullptr, 0, nullptr, o1b, 128, 8192, 128, 128, GF_GELU);
  PW(P_OUT_W2, 128 * 85);
  gemm(o1b, 128, wslot, 85, 0, F(P_OUT_B2), nullptr, 0, gsp_out, nullptr, 85, 8192, 85, 128, 0);
  gsp_adjust_kernel<<<32, 256, 0, stream>>>(gsp_out, scalars, 8192);
  PW(P_OFF_W1, 128 * 64);
  gemm(sfb, 128, wslot, 64, 0, F(P_OFF_B1), nullptr, 0, nullptr, off1b, 64, 8192, 64, 128, GF_GELU);
  PW(P_OFF_W2, 64 * 3);
  gemm(off1b, 64, wslot, 3, 0, F(P_OFF_B2), nullptr, 0, offv, nullptr, 3, 8192, 3, 64, 0);
  offsets_kernel<<<(8192 * 3 + 255) / 256, 256, 0, stream>>>(offv, all_pts, scalars, gpos_out, 8192);
}
